// LSTMClassifier_8143257994012
// MI455X (gfx1250) — matrix operand order repaired and checked by reading, not compiled, not run on hardware
//
#include <hip/hip_runtime.h>
#include <hip/hip_fp16.h>

typedef __attribute__((ext_vector_type(16))) _Float16 v16h;
typedef __attribute__((ext_vector_type(8)))  _Float16 v8h;
typedef __attribute__((ext_vector_type(4)))  _Float16 v4h;
typedef __attribute__((ext_vector_type(8)))  float    v8f;

#define T_STEPS 256
#define BATCH   64
#define IN0     512
#define HID     1024
#define NGATE   4096   // 4 * HID
#define NWG     64     // persistent workgroups (one hidden 16-block each)
#define WG_THREADS 128 // 4 waves per WG -> 256 waves total
#define LDS_PAD 8      // halfs; row stride 516 dwords == 4 mod 64 banks

// ---------------------------------------------------------------------------
// small helpers
// ---------------------------------------------------------------------------
__device__ __forceinline__ float sigmoid_fast(float x) {
    return 1.0f / (1.0f + __expf(-x));
}
__device__ __forceinline__ float tanh_fast(float x) {
    x = fminf(fmaxf(x, -15.0f), 15.0f);
    float e = __expf(2.0f * x);
    return (e - 1.0f) / (e + 1.0f);
}

__device__ __forceinline__ v8f wmma_f16(v16h a, v16h b, v8f c) {
    return __builtin_amdgcn_wmma_f32_16x16x32_f16(false, a, false, b,
                                                  (short)0, c, false, false);
}

// A-matrix fragment 16x32 f16 (ISA 7.12.2): arow is the lane-resolved row
// pointer already offset by (lane>>4)*8 halfs; halfs 0..7 = K k+0..7(+8),
// halfs 8..15 = K k+16..23(+24).
__device__ __forceinline__ v16h load_afrag(const _Float16* __restrict__ arow, int k0) {
    v8h lo = *(const v8h*)(arow + k0);
    v8h hi = *(const v8h*)(arow + k0 + 16);
    v16h a;
#pragma unroll
    for (int i = 0; i < 8; ++i) { a[i] = lo[i]; a[i + 8] = hi[i]; }
    return a;
}

// B-matrix fragment 32x16 f16 from an LDS-resident [N][K] slice: lane n holds
// K k+0..7(+8) in halfs 0..7 and K k+16..23(+24) in halfs 8..15 (lanes 16-31 +8).
__device__ __forceinline__ v16h load_bfrag(const _Float16* rowptr, int k0, int lane) {
    const _Float16* p = rowptr + k0 + ((lane >> 4) << 3);
    v8h lo = *(const v8h*)(p);
    v8h hi = *(const v8h*)(p + 16);
    v16h b;
#pragma unroll
    for (int i = 0; i < 8; ++i) { b[i] = lo[i]; b[i + 8] = hi[i]; }
    return b;
}

// ---------------------------------------------------------------------------
// pack / init kernels
// ---------------------------------------------------------------------------
__global__ void cvt_f32_to_f16_x4(const float* __restrict__ src,
                                  _Float16* __restrict__ dst, int n4) {
    int i = blockIdx.x * blockDim.x + threadIdx.x;
    if (i >= n4) return;
    float4 v = ((const float4*)src)[i];
    v4h h;
    h[0] = (_Float16)v.x; h[1] = (_Float16)v.y;
    h[2] = (_Float16)v.z; h[3] = (_Float16)v.w;
    ((v4h*)dst)[i] = h;
}

__global__ void bias_combine(const float* __restrict__ b0,
                             const float* __restrict__ b1,
                             float* __restrict__ out, int n) {
    int i = blockIdx.x * blockDim.x + threadIdx.x;
    if (i < n) out[i] = b0[i] + b1[i];
}

__global__ void init_barriers(int* bar) {
    bar[0] = 0;
    bar[1] = 0;
}

// ---------------------------------------------------------------------------
// persistent LSTM layer kernel (KX = input width: 512 for layer0, 1024 for L1)
//   xsrc : [T][BATCH][KX]  f16
//   w_ih : [NGATE][KX]     f16   (row = gate output neuron)
//   w_hh : [NGATE][HID]    f16
//   bias : [NGATE]         f32   (b_ih + b_hh)
//   hout : [T][BATCH][HID] f16
// WG b owns hidden block j0 = 16*b; its 4 waves cover batch tiles mt=0..3.
// The 64 weight rows this WG ever needs (4 gates x 16 cols, for both w_ih and
// w_hh) are staged into LDS ONCE and feed all 256 steps (ds_load_b128).
// Cell state c lives in 8 VGPRs per wave for the entire sequence.
// ---------------------------------------------------------------------------
template <int KX>
__global__ void __launch_bounds__(WG_THREADS)
lstm_layer_kernel(const _Float16* __restrict__ xsrc,
                  const _Float16* __restrict__ w_ih,
                  const _Float16* __restrict__ w_hh,
                  const float* __restrict__ bias,
                  _Float16* __restrict__ hout,
                  int* __restrict__ bar) {
    constexpr int PKX = KX + LDS_PAD;
    constexpr int PKH = HID + LDS_PAD;

    extern __shared__ _Float16 smem[];
    _Float16* wih_s = smem;                 // [64][PKX]
    _Float16* whh_s = smem + 64 * PKX;      // [64][PKH]

    const int lane = threadIdx.x & 31;
    const int wave = threadIdx.x >> 5;
    const int wid  = blockIdx.x * (WG_THREADS >> 5) + wave;  // 0..255
    const int mt   = wid & 3;            // batch tile (differs per wave)
    const int jb   = wid >> 2;           // hidden 16-block (== blockIdx.x)
    const int row0 = mt * 16;
    const int j0   = jb * 16;
    const int ncol = j0 + (lane & 15);   // this lane's hidden column

    // ---- stage this WG's weight slice into LDS (once) ----
    // LDS row r (0..63): gate g = r>>4, col c = r&15 -> global row g*HID+j0+c
    for (int idx = threadIdx.x; idx < 64 * (KX / 8); idx += WG_THREADS) {
        const int row = idx / (KX / 8);
        const int kc  = (idx % (KX / 8)) * 8;
        const size_t n = (size_t)((row >> 4) * HID + j0 + (row & 15));
        *(v8h*)(wih_s + row * PKX + kc) = *(const v8h*)(w_ih + n * KX + kc);
    }
    for (int idx = threadIdx.x; idx < 64 * (HID / 8); idx += WG_THREADS) {
        const int row = idx / (HID / 8);
        const int kc  = (idx % (HID / 8)) * 8;
        const size_t n = (size_t)((row >> 4) * HID + j0 + (row & 15));
        *(v8h*)(whh_s + row * PKH + kc) = *(const v8h*)(w_hh + n * HID + kc);
    }
    __syncthreads();

    // per-lane LDS weight row pointers + gate bias
    float biasv[4];
    const _Float16* wih_row[4];
    const _Float16* whh_row[4];
#pragma unroll
    for (int g = 0; g < 4; ++g) {
        biasv[g]   = bias[g * HID + ncol];
        wih_row[g] = wih_s + (g * 16 + (lane & 15)) * PKX;
        whh_row[g] = whh_s + (g * 16 + (lane & 15)) * PKH;
    }

    // per-lane A-row offsets (hoisted out of the k-loops)
    const int arow_m  = row0 + (lane & 15);
    const int akoff   = (lane >> 4) << 3;

    float c_state[8];
#pragma unroll
    for (int r = 0; r < 8; ++r) c_state[r] = 0.0f;

    const int nblocks = gridDim.x;

    for (int t = 0; t < T_STEPS; ++t) {
        v8f acc[4];
#pragma unroll
        for (int g = 0; g < 4; ++g)
#pragma unroll
            for (int r = 0; r < 8; ++r) acc[g][r] = biasv[g];

        // ---- input contribution: gates += x_t * w_ih^T ----
        const _Float16* arow_x =
            xsrc + (size_t)t * BATCH * KX + (size_t)arow_m * KX + akoff;
#pragma unroll 4
        for (int k0 = 0; k0 < KX; k0 += 32) {
            __builtin_prefetch(arow_x + k0 + 128, 0, 1);
            v16h a = load_afrag(arow_x, k0);
#pragma unroll
            for (int g = 0; g < 4; ++g) {
                v16h b = load_bfrag(wih_row[g], k0, lane);
                acc[g] = wmma_f16(a, b, acc[g]);
            }
        }

        // ---- recurrent contribution: gates += h_{t-1} * w_hh^T ----
        if (t > 0) {
            const _Float16* arow_h =
                hout + (size_t)(t - 1) * BATCH * HID + (size_t)arow_m * HID + akoff;
#pragma unroll 4
            for (int k0 = 0; k0 < HID; k0 += 32) {
                __builtin_prefetch(arow_h + k0 + 128, 0, 1);
                v16h a = load_afrag(arow_h, k0);
#pragma unroll
                for (int g = 0; g < 4; ++g) {
                    v16h b = load_bfrag(whh_row[g], k0, lane);
                    acc[g] = wmma_f16(a, b, acc[g]);
                }
            }
        }

        // ---- cell update (register-resident; acc0..3 = i,f,g,o aligned) ----
        _Float16* hdst = hout + (size_t)t * BATCH * HID;
#pragma unroll
        for (int r = 0; r < 8; ++r) {
            const int brow = row0 + r + ((lane >> 4) << 3);  // C/D layout (7.12.2)
            float iv = sigmoid_fast(acc[0][r]);
            float fv = sigmoid_fast(acc[1][r]);
            float gv = tanh_fast(acc[2][r]);
            float ov = sigmoid_fast(acc[3][r]);
            float cc = fv * c_state[r] + iv * gv;
            c_state[r] = cc;
            float hh = ov * tanh_fast(cc);
            hdst[(size_t)brow * HID + ncol] = (_Float16)hh;
        }

        // ---- grid-wide barrier (monotonic counter, graph-replay safe) ----
        __threadfence();
        __syncthreads();
        if (threadIdx.x == 0) {
            __hip_atomic_fetch_add(bar, 1, __ATOMIC_RELEASE, __HIP_MEMORY_SCOPE_AGENT);
            const int target = nblocks * (t + 1);
            while (__hip_atomic_load(bar, __ATOMIC_ACQUIRE, __HIP_MEMORY_SCOPE_AGENT) < target)
                __builtin_amdgcn_s_sleep(2);
        }
        __syncthreads();
        __builtin_amdgcn_fence(__ATOMIC_ACQUIRE, "agent");
    }
}

// ---------------------------------------------------------------------------
// final FC + sigmoid: one wave per (t,b) row, length-1024 dot
// ---------------------------------------------------------------------------
__global__ void fc_kernel(const _Float16* __restrict__ hin,
                          const float* __restrict__ fc_w,
                          const float* __restrict__ fc_b,
                          float* __restrict__ out) {
    const int wid  = (blockIdx.x * blockDim.x + threadIdx.x) >> 5;
    const int lane = threadIdx.x & 31;
    if (wid >= T_STEPS * BATCH) return;
    const _Float16* h = hin + (size_t)wid * HID;
    float sum = 0.0f;
#pragma unroll
    for (int i = 0; i < 4; ++i) {
        const int k = (i * 32 + lane) * 8;     // coalesced 16B per lane
        v8h hv = *(const v8h*)(h + k);
#pragma unroll
        for (int j = 0; j < 8; ++j)
            sum += (float)hv[j] * fc_w[k + j];
    }
#pragma unroll
    for (int off = 16; off > 0; off >>= 1)
        sum += __shfl_down(sum, off, 32);
    if (lane == 0)
        out[wid] = sigmoid_fast(sum + fc_b[0]);
}

// ---------------------------------------------------------------------------
// host-side launch
// ---------------------------------------------------------------------------
extern "C" void kernel_launch(void* const* d_in, const int* in_sizes, int n_in,
                              void* d_out, int out_size, void* d_ws, size_t ws_size,
                              hipStream_t stream) {
    const float* x     = (const float*)d_in[0];
    const float* w_ih0 = (const float*)d_in[1];
    const float* w_hh0 = (const float*)d_in[2];
    const float* b_ih0 = (const float*)d_in[3];
    const float* b_hh0 = (const float*)d_in[4];
    const float* w_ih1 = (const float*)d_in[5];
    const float* w_hh1 = (const float*)d_in[6];
    const float* b_ih1 = (const float*)d_in[7];
    const float* b_hh1 = (const float*)d_in[8];
    const float* fc_w  = (const float*)d_in[9];
    const float* fc_b  = (const float*)d_in[10];
    (void)in_sizes; (void)n_in; (void)out_size; (void)ws_size;

    // workspace carve-up (256B aligned slices)
    char* p = (char*)d_ws;
    auto carve = [&](size_t bytes) -> void* {
        void* r = (void*)p;
        p += (bytes + 255) & ~(size_t)255;
        return r;
    };
    _Float16* x_h    = (_Float16*)carve((size_t)T_STEPS * BATCH * IN0 * 2);
    _Float16* wih0_h = (_Float16*)carve((size_t)NGATE * IN0 * 2);
    _Float16* whh0_h = (_Float16*)carve((size_t)NGATE * HID * 2);
    _Float16* wih1_h = (_Float16*)carve((size_t)NGATE * HID * 2);
    _Float16* whh1_h = (_Float16*)carve((size_t)NGATE * HID * 2);
    float*    bias0  = (float*)carve((size_t)NGATE * 4);
    float*    bias1  = (float*)carve((size_t)NGATE * 4);
    _Float16* out0_h = (_Float16*)carve((size_t)T_STEPS * BATCH * HID * 2);
    _Float16* out1_h = (_Float16*)carve((size_t)T_STEPS * BATCH * HID * 2);
    int*      bar    = (int*)carve(256);

    auto cvt = [&](const float* src, _Float16* dst, size_t n) {
        int n4 = (int)(n / 4);
        cvt_f32_to_f16_x4<<<(n4 + 255) / 256, 256, 0, stream>>>(src, dst, n4);
    };
    cvt(x,     x_h,    (size_t)T_STEPS * BATCH * IN0);
    cvt(w_ih0, wih0_h, (size_t)NGATE * IN0);
    cvt(w_hh0, whh0_h, (size_t)NGATE * HID);
    cvt(w_ih1, wih1_h, (size_t)NGATE * HID);
    cvt(w_hh1, whh1_h, (size_t)NGATE * HID);

    bias_combine<<<NGATE / 256, 256, 0, stream>>>(b_ih0, b_hh0, bias0, NGATE);
    bias_combine<<<NGATE / 256, 256, 0, stream>>>(b_ih1, b_hh1, bias1, NGATE);
    init_barriers<<<1, 1, 0, stream>>>(bar);

    // dynamic LDS: 64 weight rows for w_ih and w_hh, padded to avoid bank
    // conflicts (199 KB layer0 / 258 KB layer1; WGP LDS is 320 KB)
    const size_t smem0 =
        (size_t)(64 * (IN0 + LDS_PAD) + 64 * (HID + LDS_PAD)) * sizeof(_Float16);
    const size_t smem1 =
        (size_t)(64 * (HID + LDS_PAD) + 64 * (HID + LDS_PAD)) * sizeof(_Float16);

    lstm_layer_kernel<IN0><<<NWG, WG_THREADS, smem0, stream>>>(
        x_h, wih0_h, whh0_h, bias0, out0_h, bar + 0);
    lstm_layer_kernel<HID><<<NWG, WG_THREADS, smem1, stream>>>(
        out0_h, wih1_h, whh1_h, bias1, out1_h, bar + 1);

    const int rows = T_STEPS * BATCH;                 // one wave per row
    fc_kernel<<<(rows * 32 + 255) / 256, 256, 0, stream>>>(out1_h, fc_w, fc_b,
                                                           (float*)d_out);
}
